// SparseGraphAttentionLayer_51333449121925
// MI455X (gfx1250) — matrix operand order repaired and checked by reading, not compiled, not run on hardware
//
#include <hip/hip_runtime.h>
#include <hip/hip_bf16.h>

#define N_NODES 100000
#define N_EDGES 1600000
#define IN_FEAT 128
#define OUT_FEAT 64
#define LRELU_ALPHA 0.2f
#define GAT_EPS 1e-16f

typedef __attribute__((ext_vector_type(16))) __bf16 bf16x16;
typedef __attribute__((ext_vector_type(8)))  float  f32x8;

// Monotonic float <-> ordered-uint mapping (for atomicMax on f32).
__device__ __forceinline__ unsigned fkey(float f) {
    unsigned u = __float_as_uint(f);
    return (u & 0x80000000u) ? ~u : (u | 0x80000000u);
}
__device__ __forceinline__ float funkey(unsigned k) {
    unsigned u = (k & 0x80000000u) ? (k ^ 0x80000000u) : ~k;
    return __uint_as_float(u);
}

// ---------------------------------------------------------------------------
// Kernel 1: Wh = x @ W via v_wmma_f32_16x16x32_bf16.
// Block = 256 threads = 8 waves; each wave computes a 16x64 tile of Wh.
// W (128x64) is staged in LDS *pre-swizzled into per-lane B fragments*:
//   Wlds[((k*4+nt)*32 + lane)*16 + j] = bf16(W[K,col]) with
//   col = nt*16 + (lane&15), K = k*32 + (lane>>4)*8 + (j&7) + (j>>3)*16
// so a B fragment is one contiguous 32B lane read (2x ds_load_b128).
// A fragments are loaded as 4x float4 (global_load_b128) and packed with
// v_cvt_pk_bf16_f32.
// ---------------------------------------------------------------------------
__global__ void gat_gemm_wmma(const float* __restrict__ x,
                              const float* __restrict__ W,
                              float* __restrict__ Wh) {
    __shared__ __attribute__((aligned(32))) __bf16 Wlds[IN_FEAT * OUT_FEAT];

    const int tid = threadIdx.x;
    // Cooperative swizzled stage of W (8192 bf16 = 16 KB).
    for (int idx = tid; idx < IN_FEAT * OUT_FEAT; idx += 256) {
        const int j    = idx & 15;
        const int frag = idx >> 4;          // (k*4+nt)*32 + lane
        const int ln   = frag & 31;
        const int nt   = (frag >> 5) & 3;
        const int k    = frag >> 7;
        const int col  = nt * 16 + (ln & 15);
        const int K    = k * 32 + (ln >> 4) * 8 + (j & 7) + (j >> 3) * 16;
        Wlds[idx] = (__bf16)W[K * OUT_FEAT + col];
    }
    __syncthreads();

    const int wave = tid >> 5;
    const int lane = tid & 31;
    const int half = lane >> 4;    // 0: lanes 0-15, 1: lanes 16-31
    const int lm   = lane & 15;

    const long rowTile = (long)blockIdx.x * 128 + (long)wave * 16;
    if (rowTile >= N_NODES) return;   // wave-uniform exit (after barrier)

    long row = rowTile + lm;
    if (row >= N_NODES) row = N_NODES - 1;   // clamp; stores are guarded

    f32x8 acc[4] = {};

#pragma unroll
    for (int k = 0; k < 4; ++k) {
        const int Kbase = k * 32;
        // A fragment: elems 0..7 -> K = Kbase + half*8 + (0..7)
        //             elems 8..15 -> K = Kbase + half*8 + 16 + (0..7)
        const float4* xp = (const float4*)(x + row * IN_FEAT + Kbase + half * 8);
        float av[16];
        *(float4*)(av + 0)  = xp[0];
        *(float4*)(av + 4)  = xp[1];
        *(float4*)(av + 8)  = xp[4];
        *(float4*)(av + 12) = xp[5];
        bf16x16 aF;
#pragma unroll
        for (int j = 0; j < 16; ++j) aF[j] = (__bf16)av[j];

#pragma unroll
        for (int nt = 0; nt < 4; ++nt) {
            const bf16x16 bF =
                *(const bf16x16*)(Wlds + (((k * 4 + nt) * 32 + lane) << 4));
            acc[nt] = __builtin_amdgcn_wmma_f32_16x16x32_bf16(
                false, aF, false, bF, (short)0, acc[nt], false, false);
        }
    }

    // C/D layout: VGPR r -> M = r + half*8, N = lm (within N-tile nt).
#pragma unroll
    for (int nt = 0; nt < 4; ++nt) {
        const int col = nt * 16 + lm;
#pragma unroll
        for (int r = 0; r < 8; ++r) {
            const long orow = rowTile + half * 8 + r;
            if (orow < N_NODES)
                Wh[orow * OUT_FEAT + col] = acc[nt][r];
        }
    }
}

// ---------------------------------------------------------------------------
// Kernel 2: per-node attention scores (Wh row dotted with a_src / a_tgt).
// float4 loads; 25.6 MB sweep, L2-resident.
// ---------------------------------------------------------------------------
__global__ void gat_scores(const float* __restrict__ Wh,
                           const float* __restrict__ a,
                           float* __restrict__ ssrc,
                           float* __restrict__ stgt) {
    const long i = (long)blockIdx.x * blockDim.x + threadIdx.x;
    if (i >= N_NODES) return;
    const float4* w  = (const float4*)(Wh + i * OUT_FEAT);
    const float4* a0 = (const float4*)a;
    const float4* a1 = (const float4*)(a + OUT_FEAT);
    float s0 = 0.f, s1 = 0.f;
#pragma unroll
    for (int c = 0; c < OUT_FEAT / 4; ++c) {
        const float4 v = w[c], p = a0[c], q = a1[c];
        s0 += v.x * p.x + v.y * p.y + v.z * p.z + v.w * p.w;
        s1 += v.x * q.x + v.y * q.y + v.z * q.z + v.w * q.w;
    }
    ssrc[i] = s0;
    stgt[i] = s1;
}

// ---------------------------------------------------------------------------
// Kernel 3: init accumulators (d_out / d_ws are poisoned by the harness).
// ---------------------------------------------------------------------------
__global__ void gat_init(unsigned* __restrict__ emax,
                         float* __restrict__ esum,
                         float* __restrict__ hprime) {
    const long i = (long)blockIdx.x * blockDim.x + threadIdx.x;
    if (i < (long)N_NODES * OUT_FEAT) hprime[i] = 0.f;
    if (i < N_NODES) {
        emax[i] = 0x007FFFFFu;   // fkey(-inf)
        esum[i] = 0.f;
    }
}

// ---------------------------------------------------------------------------
// Kernel 4: per-edge leaky-relu score + segment max (ordered-uint atomicMax).
// ---------------------------------------------------------------------------
__global__ void gat_edge_max(const long long* __restrict__ ei,
                             const float* __restrict__ ssrc,
                             const float* __restrict__ stgt,
                             float* __restrict__ ebuf,
                             unsigned* __restrict__ emax) {
    const long i = (long)blockIdx.x * blockDim.x + threadIdx.x;
    if (i >= N_EDGES) return;
    const long s = (long)ei[i];
    const long t = (long)ei[N_EDGES + i];
    float e = ssrc[s] + stgt[t];
    e = (e > 0.f) ? e : LRELU_ALPHA * e;
    ebuf[i] = e;
    atomicMax(emax + t, fkey(e));
}

// ---------------------------------------------------------------------------
// Kernel 5: per-edge exp(e - max) + segment sum.
// ---------------------------------------------------------------------------
__global__ void gat_edge_exp(const long long* __restrict__ ei,
                             const float* __restrict__ ebuf,
                             const unsigned* __restrict__ emax,
                             float* __restrict__ eexp,
                             float* __restrict__ esum) {
    const long i = (long)blockIdx.x * blockDim.x + threadIdx.x;
    if (i >= N_EDGES) return;
    const long t = (long)ei[N_EDGES + i];
    const float ee = expf(ebuf[i] - funkey(emax[t]));
    eexp[i] = ee;
    atomicAdd(esum + t, ee);
}

// ---------------------------------------------------------------------------
// Kernel 5b: per-edge normalized attention weight (overwrites ebuf).
// Precomputing alpha removes the e_sum gather + divide from the hot scatter.
// ---------------------------------------------------------------------------
__global__ void gat_alpha(const long long* __restrict__ ei,
                          const float* __restrict__ eexp,
                          const float* __restrict__ esum,
                          float* __restrict__ alpha) {
    const long i = (long)blockIdx.x * blockDim.x + threadIdx.x;
    if (i >= N_EDGES) return;
    const long t = (long)ei[N_EDGES + i];
    alpha[i] = eexp[i] / (esum[t] + GAT_EPS);
}

// ---------------------------------------------------------------------------
// Kernel 6: weighted scatter-add: h'[tgt] += alpha * Wh[src].
// One thread per (edge, feature): 102.4M global_atomic_add_f32, all
// L2-resident (out = 25.6 MB). This is the roofline-dominant stage.
// ---------------------------------------------------------------------------
__global__ void gat_scatter(const long long* __restrict__ ei,
                            const float* __restrict__ Wh,
                            const float* __restrict__ alpha,
                            float* __restrict__ hprime) {
    const size_t idx = (size_t)blockIdx.x * blockDim.x + threadIdx.x;
    if (idx >= (size_t)N_EDGES * OUT_FEAT) return;
    const long e = (long)(idx >> 6);
    const int  f = (int)(idx & 63);
    const long s = (long)ei[e];
    const long t = (long)ei[N_EDGES + e];
    atomicAdd(hprime + t * OUT_FEAT + f, Wh[s * OUT_FEAT + f] * alpha[e]);
}

// ---------------------------------------------------------------------------
// Kernel 7: ELU in place on the accumulated output (float4 sweep).
// ---------------------------------------------------------------------------
__global__ void gat_elu(float* __restrict__ h) {
    const long i = (long)blockIdx.x * blockDim.x + threadIdx.x;
    if (i >= (long)(N_NODES * OUT_FEAT / 4)) return;
    float4 v = ((float4*)h)[i];
    v.x = (v.x > 0.f) ? v.x : (expf(v.x) - 1.f);
    v.y = (v.y > 0.f) ? v.y : (expf(v.y) - 1.f);
    v.z = (v.z > 0.f) ? v.z : (expf(v.z) - 1.f);
    v.w = (v.w > 0.f) ? v.w : (expf(v.w) - 1.f);
    ((float4*)h)[i] = v;
}

// ---------------------------------------------------------------------------
extern "C" void kernel_launch(void* const* d_in, const int* in_sizes, int n_in,
                              void* d_out, int out_size, void* d_ws, size_t ws_size,
                              hipStream_t stream) {
    const float*     x  = (const float*)d_in[0];
    const long long* ei = (const long long*)d_in[1];   // int64 edge_index (2, E)
    const float*     W  = (const float*)d_in[2];
    const float*     a  = (const float*)d_in[3];
    float* out = (float*)d_out;                        // N x OUT_FEAT (h' then ELU)

    // Workspace layout (256B aligned regions).
    char* ws = (char*)d_ws;
    size_t off = 0;
    auto alloc = [&](size_t bytes) {
        char* p = ws + off;
        off = (off + bytes + 255) & ~(size_t)255;
        return p;
    };
    float*    Wh   = (float*)   alloc((size_t)N_NODES * OUT_FEAT * sizeof(float));
    float*    ssrc = (float*)   alloc((size_t)N_NODES * sizeof(float));
    float*    stgt = (float*)   alloc((size_t)N_NODES * sizeof(float));
    unsigned* emax = (unsigned*)alloc((size_t)N_NODES * sizeof(unsigned));
    float*    esum = (float*)   alloc((size_t)N_NODES * sizeof(float));
    float*    ebuf = (float*)   alloc((size_t)N_EDGES * sizeof(float));  // e, then alpha
    float*    eexp = (float*)   alloc((size_t)N_EDGES * sizeof(float));
    (void)ws_size; (void)in_sizes; (void)n_in; (void)out_size;

    const int B = 256;
    const int gGemm = (N_NODES + 127) / 128;                            // 782
    const int gNode = (N_NODES + B - 1) / B;                            // 391
    const int gNF   = (int)(((long)N_NODES * OUT_FEAT + B - 1) / B);    // 25000
    const int gNF4  = (int)(((long)N_NODES * OUT_FEAT / 4 + B - 1) / B);// 6250
    const int gEdge = (N_EDGES + B - 1) / B;                            // 6250
    const int gEF   = (int)(((size_t)N_EDGES * OUT_FEAT + B - 1) / B);  // 400000

    gat_gemm_wmma<<<gGemm, B, 0, stream>>>(x, W, Wh);
    gat_scores  <<<gNode, B, 0, stream>>>(Wh, a, ssrc, stgt);
    gat_init    <<<gNF,   B, 0, stream>>>(emax, esum, out);
    gat_edge_max<<<gEdge, B, 0, stream>>>(ei, ssrc, stgt, ebuf, emax);
    gat_edge_exp<<<gEdge, B, 0, stream>>>(ei, ebuf, emax, eexp, esum);
    gat_alpha   <<<gEdge, B, 0, stream>>>(ei, eexp, esum, ebuf);
    gat_scatter <<<gEF,   B, 0, stream>>>(ei, Wh, ebuf, out);
    gat_elu     <<<gNF4,  B, 0, stream>>>(out);
}
